// AttentionBlock_37830071943278
// MI455X (gfx1250) — compile-verified, matrix operand order checked by reading, not run on hardware
//
#include <hip/hip_runtime.h>

// ---------------------------------------------------------------------------
// CDNA5 (gfx1250, wave32) fused BERT-attention forward.
// - All GEMMs via v_wmma_f32_16x16x32_bf16
// - Flash-style 2-pass softmax: 402MB probs written exactly once (NT stores)
// - K/V tiles staged to LDS with global_load_async_to_lds_b128 (ASYNCcnt),
//   double-buffered so the copy of chunk k+1 overlaps the WMMA of chunk k.
// - Dense GEMM kernels (qkv, out-proj) use explicit ping-pong register
//   buffers (2x unrolled) so WMMAs overlap loads with zero register moves.
// ---------------------------------------------------------------------------

typedef __attribute__((ext_vector_type(16))) __bf16 v16bf;
typedef __attribute__((ext_vector_type(8)))  float  v8f;

constexpr int   kB   = 2;
constexpr int   kS   = 2048;
constexpr int   kH   = 768;
constexpr int   kNH  = 12;
constexpr int   kHD  = 64;
constexpr int   kBS  = kB * kS;          // 4096 rows total
constexpr float kGamma = 0.05f;
constexpr float kEps   = 1e-12f;

// workspace byte offsets (total ~36.6 MB)
constexpr size_t OFF_XB  = 0;                         // x bf16        [4096][768]
constexpr size_t OFF_WQ  = 6291456;                   // Wq bf16       [768][768]
constexpr size_t OFF_WK  = 7471104;
constexpr size_t OFF_WV  = 8650752;
constexpr size_t OFF_WO  = 9830400;                   // gamma-mod Wo bf16
constexpr size_t OFF_PBO = 11010048;                  // gamma-mod bo f32 [768]
constexpr size_t OFF_QB  = 11013120;                  // q bf16        [4096][768]
constexpr size_t OFF_KB  = 17304576;                  // k bf16        [4096][768]
constexpr size_t OFF_VT  = 23596032;                  // vT bf16       [2][12][64][2048]
constexpr size_t OFF_CTX = 29887488;                  // ctx bf16      [4096][768]
constexpr size_t OFF_M   = 36178944;                  // rowmax f32    [2][12][2048]
constexpr size_t OFF_L   = 36375552;                  // rowsum f32    [2][12][2048]

union C8 { v8f v; float f[8]; };
struct __align__(32) Pair16 { uint4 lo, hi; };

// Load one 16x32 bf16 WMMA fragment (A or B) from a row-major matrix
// (global or LDS). p must point at (row = tileRow + (lane&15), col = kbase).
// Lane layout (ISA 7.12.2): lanes 0-15 hold K 0-7 & 16-23, lanes 16-31 hold
// K 8-15 & 24-31 -> two 16B loads at byte offsets 16*kh and 32+16*kh.
static __device__ __forceinline__ v16bf frag_rm(const __bf16* p, int kh) {
  Pair16 pr;
  pr.lo = *(const uint4*)((const char*)p + 16 * kh);
  pr.hi = *(const uint4*)((const char*)p + 32 + 16 * kh);
  return __builtin_bit_cast(v16bf, pr);
}

static __device__ __forceinline__ v8f wmma_bf16(v16bf a, v16bf b, v8f c) {
  return __builtin_amdgcn_wmma_f32_16x16x32_bf16(false, a, false, b,
                                                 (short)0, c, false, false);
}

static __device__ __forceinline__ v8f zero8() {
  v8f z = {0.f, 0.f, 0.f, 0.f, 0.f, 0.f, 0.f, 0.f};
  return z;
}

// --- CDNA5 async global->LDS copy (16 bytes per lane), tracked by ASYNCcnt --
// the LDS address is taken from the vdst VGPR; generic LDS pointers carry the LDS offset in
// their low 32 bits, so truncation yields the required VGPR value.
static __device__ __forceinline__ void async_cp16(const void* g, void* l) {
  unsigned lofs = (unsigned)(size_t)l;
  unsigned long long ga = (unsigned long long)(size_t)g;
  asm volatile("global_load_async_to_lds_b128 %0, %1, off"
               :: "v"(lofs), "v"(ga) : "memory");
}

static __device__ __forceinline__ void wait_async0() {
  asm volatile("s_wait_asynccnt 0x0" ::: "memory");
}

// ---------------- packing / precision conversion kernels -------------------

__global__ void k_cvt_bf16(const float* __restrict__ src,
                           __bf16* __restrict__ dst, int n) {
  int i = blockIdx.x * blockDim.x + threadIdx.x;
  if (i < n) dst[i] = (__bf16)src[i];
}

__global__ void k_cvt_gamma_bf16(const float* __restrict__ src,
                                 __bf16* __restrict__ dst, int n) {
  int i = blockIdx.x * blockDim.x + threadIdx.x;
  if (i < n) {
    float w = src[i];
    dst[i] = (__bf16)(w + kGamma * fmaxf(w, 0.f));
  }
}

__global__ void k_gamma_f32(const float* __restrict__ src,
                            float* __restrict__ dst, int n) {
  int i = blockIdx.x * blockDim.x + threadIdx.x;
  if (i < n) {
    float w = src[i];
    dst[i] = w + kGamma * fmaxf(w, 0.f);
  }
}

// ---------------- QKV projection: y = x @ W.T + b ---------------------------
// grid (128, 3, 3) block 256. Each wave: 16 rows x 64 cols, K=768.
// Ping-pong register buffers (2x unrolled): set B loads during set-A WMMAs
// and vice versa -> pipelined loads with no register rotation moves.
// z selects q/k/v. v is stored transposed per head -> vT[b][h][d][s].

__global__ void k_qkv(const __bf16* __restrict__ xb,
                      const __bf16* __restrict__ wq, const __bf16* __restrict__ wk,
                      const __bf16* __restrict__ wv,
                      const float* __restrict__ bq, const float* __restrict__ bk,
                      const float* __restrict__ bv,
                      __bf16* __restrict__ qb, __bf16* __restrict__ kb,
                      __bf16* __restrict__ vT) {
  const int z = blockIdx.z;
  const __bf16* wb  = (z == 0) ? wq : (z == 1) ? wk : wv;
  const float* bias = (z == 0) ? bq : (z == 1) ? bk : bv;

  const int wave = threadIdx.x >> 5, lane = threadIdx.x & 31;
  const int rl = lane & 15, kh = lane >> 4;
  const int rbase = blockIdx.x * 32 + (wave & 1) * 16;
  const int cbase = blockIdx.y * 256 + (wave >> 1) * 64;

  const __bf16* arow = xb + (size_t)(rbase + rl) * kH;
  const __bf16* brow[4];
  for (int nt = 0; nt < 4; ++nt)
    brow[nt] = wb + (size_t)(cbase + nt * 16 + rl) * kH;

  C8 acc[4];
  for (int t = 0; t < 4; ++t) acc[t].v = zero8();

  constexpr int NK = kH / 32;                      // 24 (even)
  v16bf aA = frag_rm(arow, kh);
  v16bf bA[4];
  for (int nt = 0; nt < 4; ++nt) bA[nt] = frag_rm(brow[nt], kh);

  for (int kc = 0; kc < NK; kc += 2) {
    const int k1 = kc + 1;                         // always < NK
    v16bf aB = frag_rm(arow + k1 * 32, kh);
    v16bf bB[4];
    for (int nt = 0; nt < 4; ++nt) bB[nt] = frag_rm(brow[nt] + k1 * 32, kh);

    for (int nt = 0; nt < 4; ++nt)
      acc[nt].v = wmma_bf16(aA, bA[nt], acc[nt].v);

    const int k2 = (kc + 2 < NK) ? kc + 2 : 0;     // wrap: harmless dead load
    aA = frag_rm(arow + k2 * 32, kh);
    for (int nt = 0; nt < 4; ++nt) bA[nt] = frag_rm(brow[nt] + k2 * 32, kh);

    for (int nt = 0; nt < 4; ++nt)
      acc[nt].v = wmma_bf16(aB, bB[nt], acc[nt].v);
  }

  for (int nt = 0; nt < 4; ++nt) {
    for (int v = 0; v < 8; ++v) {
      const int m = v + 8 * kh;               // C/D layout: M = vgpr + 8*half
      const int n = cbase + nt * 16 + rl;     // N = lane&15
      const int row = rbase + m;
      const float val = acc[nt].f[v] + bias[n];
      if (z == 0) {
        qb[(size_t)row * kH + n] = (__bf16)val;
      } else if (z == 1) {
        kb[(size_t)row * kH + n] = (__bf16)val;
      } else {
        const int h = n >> 6, d = n & 63;
        const int b_ = row >> 11, s = row & 2047;
        vT[(((size_t)(b_ * kNH + h)) * kHD + d) * kS + s] = (__bf16)val;
      }
    }
  }
}

// ---------------- attention pass 1: flash (stats + ctx) ---------------------
// grid (32, 12, 2) block 128. WG = 64 q rows of one (b,h); wave = 16 rows.
// K/V tiles (32 keys) staged to LDS via async copies, double-buffered.

__global__ void k_attn1(const __bf16* __restrict__ qb, const __bf16* __restrict__ kb,
                        const __bf16* __restrict__ vT, __bf16* __restrict__ ctxb,
                        float* __restrict__ mbuf, float* __restrict__ lbuf) {
  const int b = blockIdx.z, h = blockIdx.y;
  const int tid = threadIdx.x;
  const int wave = tid >> 5, lane = tid & 31;
  const int rl = lane & 15, kh = lane >> 4;
  const int qrow0 = blockIdx.x * 64 + wave * 16;         // row within sequence
  const size_t qgrow = (size_t)b * kS + qrow0;           // global row in [B*S]

  __shared__ alignas(128) __bf16 ktile[2][32][64];       // 8 KB (2 x 4KB)
  __shared__ alignas(128) __bf16 vtile[2][64][32];       // 8 KB
  __shared__ alignas(64)  __bf16 plds[4][16][32];        // P tile per wave

  const __bf16* kbp = kb + (size_t)b * kS * kH + h * kHD;       // ld = kH
  const __bf16* vbp = vT + ((size_t)(b * kNH + h)) * kHD * kS;  // ld = kS

  // async-stage one 32-key chunk into buffer bf (K: 32x128B, V: 64x64B)
  auto stage = [&](int kt, int bf) {
    for (int i = 0; i < 2; ++i) {
      const int seg = tid + i * 128;                     // 0..255
      const int r = seg >> 3, s = seg & 7;               // K: 8 x 16B per row
      async_cp16((const char*)(kbp + (size_t)(kt * 32 + r) * kH) + s * 16,
                 (char*)&ktile[bf][r][0] + s * 16);
    }
    for (int i = 0; i < 2; ++i) {
      const int seg = tid + i * 128;
      const int r = seg >> 2, s = seg & 3;               // V: 4 x 16B per row
      async_cp16((const char*)(vbp + (size_t)r * kS + kt * 32) + s * 16,
                 (char*)&vtile[bf][r][0] + s * 16);
    }
  };

  stage(0, 0);

  // Q fragments load from global while the first copies are in flight
  const __bf16* qptr = qb + (qgrow + rl) * kH + h * kHD;
  const v16bf aq0 = frag_rm(qptr, kh);                   // K dims 0..31
  const v16bf aq1 = frag_rm(qptr + 32, kh);              // K dims 32..63

  float mrun[8], lrun[8];
  C8 acc[4];
  for (int v = 0; v < 8; ++v) { mrun[v] = -3.0e38f; lrun[v] = 0.f; }
  for (int t = 0; t < 4; ++t) acc[t].v = zero8();

  wait_async0();
  __syncthreads();

  for (int kt = 0; kt < kS / 32; ++kt) {
    const int cur = kt & 1;
    if (kt + 1 < kS / 32) stage(kt + 1, cur ^ 1);        // overlap copy w/ math

    C8 s0, s1;
    {
      const __bf16* kp = &ktile[cur][rl][0];             // keys 0..15 of chunk
      v8f c = zero8();
      c = wmma_bf16(aq0, frag_rm(kp, kh), c);
      c = wmma_bf16(aq1, frag_rm(kp + 32, kh), c);
      s0.v = c;
    }
    {
      const __bf16* kp = &ktile[cur][16 + rl][0];        // keys 16..31
      v8f c = zero8();
      c = wmma_bf16(aq0, frag_rm(kp, kh), c);
      c = wmma_bf16(aq1, frag_rm(kp + 32, kh), c);
      s1.v = c;
    }

    float scale[8];
    for (int v = 0; v < 8; ++v) {
      float t = fmaxf(s0.f[v], s1.f[v]);                 // row max, 16-lane half
      t = fmaxf(t, __shfl_xor(t, 1));
      t = fmaxf(t, __shfl_xor(t, 2));
      t = fmaxf(t, __shfl_xor(t, 4));
      t = fmaxf(t, __shfl_xor(t, 8));
      const float mn = fmaxf(mrun[v], t);
      const float sc = __expf(mrun[v] - mn);
      const float p0 = __expf(s0.f[v] - mn);
      const float p1 = __expf(s1.f[v] - mn);
      plds[wave][v + 8 * kh][rl]      = (__bf16)p0;      // C-layout -> LDS
      plds[wave][v + 8 * kh][16 + rl] = (__bf16)p1;
      float ps = p0 + p1;
      ps += __shfl_xor(ps, 1);
      ps += __shfl_xor(ps, 2);
      ps += __shfl_xor(ps, 4);
      ps += __shfl_xor(ps, 8);
      lrun[v] = lrun[v] * sc + ps;
      mrun[v] = mn;
      scale[v] = sc;
    }

    // reload P as an A fragment (16x32 over keys); DScnt orders within wave
    const v16bf ap = frag_rm(&plds[wave][rl][0], kh);
    for (int t = 0; t < 4; ++t) {
      v16bf bv = frag_rm(&vtile[cur][t * 16 + rl][0], kh);
      for (int v = 0; v < 8; ++v) acc[t].f[v] *= scale[v];
      acc[t].v = wmma_bf16(ap, bv, acc[t].v);
    }

    wait_async0();        // our chunk-(kt+1) copies have landed
    __syncthreads();      // ...and everyone else's too; buf reads done
  }

  float inv[8];
  for (int v = 0; v < 8; ++v) inv[v] = 1.f / lrun[v];
  for (int t = 0; t < 4; ++t)
    for (int v = 0; v < 8; ++v) {
      const int m = v + 8 * kh;
      ctxb[(qgrow + m) * kH + h * kHD + t * 16 + rl] = (__bf16)(acc[t].f[v] * inv[v]);
    }

  if (rl == 0) {
    const size_t sb = ((size_t)(b * kNH + h)) * kS + qrow0 + 8 * kh;
    for (int v = 0; v < 8; ++v) { mbuf[sb + v] = mrun[v]; lbuf[sb + v] = lrun[v]; }
  }
}

// ---------------- attention pass 2: recompute scores, emit probs ------------
// grid (128, 12, 2) block 128; blockIdx.x = {qtile(5b), key-quarter(2b)}.
// K tiles staged via async LDS copies; probs stream uses non-temporal stores
// so the 402MB stream does not thrash L2 (K/Q stay resident).

__global__ void k_attn2(const __bf16* __restrict__ qb, const __bf16* __restrict__ kb,
                        const float* __restrict__ mbuf, const float* __restrict__ lbuf,
                        float* __restrict__ probs) {
  const int b = blockIdx.z, h = blockIdx.y;
  const int qt = blockIdx.x >> 2;
  const int kq = blockIdx.x & 3;
  const int tid = threadIdx.x;
  const int wave = tid >> 5, lane = tid & 31;
  const int rl = lane & 15, kh = lane >> 4;
  const int qrow0 = qt * 64 + wave * 16;
  const size_t qgrow = (size_t)b * kS + qrow0;

  __shared__ alignas(128) __bf16 ktile[2][16][64];       // 2 x 2KB

  const __bf16* kbp = kb + (size_t)b * kS * kH + h * kHD;

  auto stage = [&](int ktg, int bf) {                    // 16 rows x 8 segs
    const int r = tid >> 3, s = tid & 7;
    async_cp16((const char*)(kbp + (size_t)(ktg * 16 + r) * kH) + s * 16,
               (char*)&ktile[bf][r][0] + s * 16);
  };

  stage(kq * 32, 0);

  const __bf16* qptr = qb + (qgrow + rl) * kH + h * kHD;
  const v16bf aq0 = frag_rm(qptr, kh);
  const v16bf aq1 = frag_rm(qptr + 32, kh);

  float mr[8], li[8];
  const size_t sb = ((size_t)(b * kNH + h)) * kS + qrow0;
  for (int v = 0; v < 8; ++v) {
    mr[v] = mbuf[sb + v + 8 * kh];
    li[v] = 1.f / lbuf[sb + v + 8 * kh];
  }

  float* prow = probs + (((size_t)(b * kNH + h)) * kS + qrow0) * kS;

  wait_async0();
  __syncthreads();

  for (int i = 0; i < 32; ++i) {                         // 16-key tiles
    const int cur = i & 1;
    const int ktg = kq * 32 + i;
    if (i + 1 < 32) stage(ktg + 1, cur ^ 1);

    const __bf16* kp = &ktile[cur][rl][0];
    v8f c = zero8();
    c = wmma_bf16(aq0, frag_rm(kp, kh), c);
    c = wmma_bf16(aq1, frag_rm(kp + 32, kh), c);
    C8 s; s.v = c;
    for (int v = 0; v < 8; ++v) {
      const int m = v + 8 * kh;
      __builtin_nontemporal_store(__expf(s.f[v] - mr[v]) * li[v],
                                  &prow[(size_t)m * kS + ktg * 16 + rl]);
    }

    wait_async0();
    __syncthreads();
  }
}

// ---------------- output projection + residual + detached LayerNorm ---------
// grid (256) block 256; WG owns 16 rows; wave covers 96 output cols.
// Ping-pong register buffers like k_qkv.

__global__ void k_out_ln(const __bf16* __restrict__ ctxb, const __bf16* __restrict__ wo,
                         const float* __restrict__ pbo, const float* __restrict__ x,
                         float* __restrict__ out) {
  const int wave = threadIdx.x >> 5, lane = threadIdx.x & 31;
  const int rl = lane & 15, kh = lane >> 4;
  const int rbase = blockIdx.x * 16;
  const int nbase = wave * 96;

  __shared__ alignas(16) float ylds[16][kH];
  __shared__ float psum[16][16], psq[16][16];
  __shared__ float rmean[16], rinv[16];

  const __bf16* arow = ctxb + (size_t)(rbase + rl) * kH;
  const __bf16* brow[6];
  for (int nt = 0; nt < 6; ++nt)
    brow[nt] = wo + (size_t)(nbase + nt * 16 + rl) * kH;

  C8 acc[6];
  for (int t = 0; t < 6; ++t) acc[t].v = zero8();

  constexpr int NK = kH / 32;                      // 24 (even)
  v16bf aA = frag_rm(arow, kh);
  v16bf bA[6];
  for (int nt = 0; nt < 6; ++nt) bA[nt] = frag_rm(brow[nt], kh);

  for (int kc = 0; kc < NK; kc += 2) {
    const int k1 = kc + 1;
    v16bf aB = frag_rm(arow + k1 * 32, kh);
    v16bf bB[6];
    for (int nt = 0; nt < 6; ++nt) bB[nt] = frag_rm(brow[nt] + k1 * 32, kh);

    for (int nt = 0; nt < 6; ++nt)
      acc[nt].v = wmma_bf16(aA, bA[nt], acc[nt].v);

    const int k2 = (kc + 2 < NK) ? kc + 2 : 0;
    aA = frag_rm(arow + k2 * 32, kh);
    for (int nt = 0; nt < 6; ++nt) bA[nt] = frag_rm(brow[nt] + k2 * 32, kh);

    for (int nt = 0; nt < 6; ++nt)
      acc[nt].v = wmma_bf16(aB, bB[nt], acc[nt].v);
  }

  for (int nt = 0; nt < 6; ++nt)
    for (int v = 0; v < 8; ++v) {
      const int m = v + 8 * kh;
      const int n = nbase + nt * 16 + rl;
      ylds[m][n] = acc[nt].f[v] + pbo[n] + x[(size_t)(rbase + m) * kH + n];
    }
  __syncthreads();

  const int tid = threadIdx.x;
  const int r = tid & 15, c = tid >> 4;
  float s = 0.f, sq = 0.f;
  for (int j = c; j < kH; j += 16) {
    const float yv = ylds[r][j];
    s += yv; sq += yv * yv;
  }
  psum[r][c] = s; psq[r][c] = sq;
  __syncthreads();

  if (tid < 16) {
    float S_ = 0.f, Q_ = 0.f;
    for (int j = 0; j < 16; ++j) { S_ += psum[tid][j]; Q_ += psq[tid][j]; }
    const float mean = S_ / (float)kH;
    float var = (Q_ - (float)kH * mean * mean) / (float)(kH - 1);  // ddof=1
    var = fmaxf(var, 0.f);
    rmean[tid] = mean;
    rinv[tid] = 1.f / (sqrtf(var) + kEps);
  }
  __syncthreads();

  for (int j = c; j < kH; j += 16)
    __builtin_nontemporal_store((ylds[r][j] - rmean[r]) * rinv[r],
                                &out[(size_t)(rbase + r) * kH + j]);
}

// ---------------------------------------------------------------------------

extern "C" void kernel_launch(void* const* d_in, const int* in_sizes, int n_in,
                              void* d_out, int out_size, void* d_ws, size_t ws_size,
                              hipStream_t stream) {
  (void)in_sizes; (void)n_in; (void)out_size; (void)ws_size;

  const float* x  = (const float*)d_in[0];
  const float* Wq = (const float*)d_in[1];
  const float* bq = (const float*)d_in[2];
  const float* Wk = (const float*)d_in[3];
  const float* bk = (const float*)d_in[4];
  const float* Wv = (const float*)d_in[5];
  const float* bv = (const float*)d_in[6];
  const float* Wo = (const float*)d_in[7];
  const float* bo = (const float*)d_in[8];

  char* ws = (char*)d_ws;
  __bf16* xb   = (__bf16*)(ws + OFF_XB);
  __bf16* wqb  = (__bf16*)(ws + OFF_WQ);
  __bf16* wkb  = (__bf16*)(ws + OFF_WK);
  __bf16* wvb  = (__bf16*)(ws + OFF_WV);
  __bf16* wob  = (__bf16*)(ws + OFF_WO);
  float*  pbo  = (float*)(ws + OFF_PBO);
  __bf16* qb   = (__bf16*)(ws + OFF_QB);
  __bf16* kbuf = (__bf16*)(ws + OFF_KB);
  __bf16* vT   = (__bf16*)(ws + OFF_VT);
  __bf16* ctxb = (__bf16*)(ws + OFF_CTX);
  float*  mbuf = (float*)(ws + OFF_M);
  float*  lbuf = (float*)(ws + OFF_L);

  float* out   = (float*)d_out;
  float* probs = out + (size_t)kBS * kH;   // outputs concatenated: (out, probs)

  const int nX = kBS * kH, nW = kH * kH;
  k_cvt_bf16<<<(nX + 255) / 256, 256, 0, stream>>>(x, xb, nX);
  k_cvt_bf16<<<(nW + 255) / 256, 256, 0, stream>>>(Wq, wqb, nW);
  k_cvt_bf16<<<(nW + 255) / 256, 256, 0, stream>>>(Wk, wkb, nW);
  k_cvt_bf16<<<(nW + 255) / 256, 256, 0, stream>>>(Wv, wvb, nW);
  k_cvt_gamma_bf16<<<(nW + 255) / 256, 256, 0, stream>>>(Wo, wob, nW);
  k_gamma_f32<<<(kH + 255) / 256, 256, 0, stream>>>(bo, pbo, kH);

  k_qkv<<<dim3(kBS / 32, kH / 256, 3), 256, 0, stream>>>(
      xb, wqb, wkb, wvb, bq, bk, bv, qb, kbuf, vT);

  k_attn1<<<dim3(kS / 64, kNH, kB), 128, 0, stream>>>(
      qb, kbuf, vT, ctxb, mbuf, lbuf);

  k_attn2<<<dim3((kS / 64) * 4, kNH, kB), 128, 0, stream>>>(
      qb, kbuf, mbuf, lbuf, probs);

  k_out_ln<<<dim3(kBS / 16), 256, 0, stream>>>(ctxb, wob, pbo, x, out);
}
